// Distance2logprob_78700980731974
// MI455X (gfx1250) — matrix operand order repaired and checked by reading, not compiled, not run on hardware
//
#include <hip/hip_runtime.h>
#include <hip/hip_bf16.h>
#include <math.h>

typedef __attribute__((ext_vector_type(16))) _Float16 v16h;
typedef __attribute__((ext_vector_type(8)))  _Float16 v8h;
typedef __attribute__((ext_vector_type(8)))  float    v8f;
typedef __attribute__((ext_vector_type(4)))  float    v4f;

#define D_DIM    128
#define M_CHUNKS 32

// ---------------------------------------------------------------------------
// Kernel 1: f32 -> f16 row copy + row sum-of-squares (one 128-thread WG/row)
// ---------------------------------------------------------------------------
__global__ __launch_bounds__(128) void prep_rows_kernel(
    const float* __restrict__ src, _Float16* __restrict__ dst,
    float* __restrict__ sq)
{
    const unsigned row = blockIdx.x;
    const unsigned t   = threadIdx.x;             // 0..127, D_DIM threads
    const unsigned idx = row * (unsigned)D_DIM + t;
    const float x = src[idx];
    dst[idx] = (_Float16)x;

    float s = x * x;
    #pragma unroll
    for (int sh = 1; sh < 32; sh <<= 1) s += __shfl_xor(s, sh, 32);

    __shared__ float ws[4];
    if ((t & 31) == 0) ws[t >> 5] = s;
    __syncthreads();
    if (t == 0) sq[row] = ws[0] + ws[1] + ws[2] + ws[3];
}

// ---------------------------------------------------------------------------
// Kernel 2: WMMA distance tile kernel, software-pipelined.
// Workgroup = 8 waves; owns a 16-row block and an M/M_CHUNKS column chunk.
// Per iteration each wave computes a 16x16 tile (K=128 -> 4 WMMAs) while the
// next iteration's B fragments and this iteration's mask dwords are in flight.
// MCONST != 0 bakes the row stride in so v*M*4 folds into the 24-bit IOFFSET
// (one saddr+voffset address for all 8 mask loads / 8 stores).
// ---------------------------------------------------------------------------
template <int MCONST>
__global__ __launch_bounds__(256) void dist_gemm_kernel(
    const _Float16* __restrict__ E16, const _Float16* __restrict__ R16,
    const float* __restrict__ sqe,    const float* __restrict__ sqr,
    const int*  __restrict__ mask,    float* __restrict__ tout,
    float* __restrict__ Zpart, int Mrt, int n_blocks, int m_cols_per_chunk)
{
    const unsigned rstep = MCONST ? (unsigned)MCONST : (unsigned)Mrt;

    const int nb   = blockIdx.x % n_blocks;
    const int mc   = blockIdx.x / n_blocks;
    const int n0   = nb * 16;
    const int wave = threadIdx.x >> 5;            // 0..7
    const int lane = threadIdx.x & 31;
    const int lo   = lane & 15;
    const int hi   = lane >> 4;

    // --- A fragments (16 rows x 128 k), in VGPRs for the whole m-sweep.
    // Contiguous k order (k = 16*hi + s, same as B) -> two contiguous b128 loads per lane per chunk.
    v16h a[4];
    {
        const unsigned aoff = ((unsigned)(n0 + lo) << 7) + (unsigned)(hi * 16);
        #pragma unroll
        for (int kc = 0; kc < 4; ++kc) {
            v8h p0 = *(const v8h*)(E16 + aoff + kc * 32);
            v8h p1 = *(const v8h*)(E16 + aoff + kc * 32 + 8);
            a[kc] = __builtin_shufflevector(p0, p1,
                        0,1,2,3,4,5,6,7,8,9,10,11,12,13,14,15);
        }
    }

    float se[8];
    #pragma unroll
    for (int v = 0; v < 8; ++v) se[v] = sqe[n0 + hi * 8 + v];

    float zacc[8];
    #pragma unroll
    for (int v = 0; v < 8; ++v) zacc[v] = 0.0f;

    const int mbase = mc * m_cols_per_chunk;
    const int iters = m_cols_per_chunk >> 7;      // 128 cols per WG iteration
    const int mcol0 = mbase + wave * 16 + lo;

    // 32-bit divergent offsets over uniform bases (max < 2^27 elements)
    unsigned boff = ((unsigned)mcol0 << 7) + (unsigned)(hi * 16); // halves
    unsigned goff = (unsigned)(n0 + hi * 8) * rstep + (unsigned)mcol0;
    unsigned soff = (unsigned)mcol0;

    // Prologue: B fragments for iteration 0 (4 chunks x 32 contiguous halves)
    v16h b0[4];
    #pragma unroll
    for (int kc = 0; kc < 4; ++kc)
        b0[kc] = *(const v16h*)(R16 + boff + kc * 32);

    for (int it = 0; it < iters; ++it) {
        const bool has_next = (it + 1 < iters);
        const unsigned boff_n = boff + (128u << 7);   // +128 R-rows of halves

        // Prefetch next iteration's B fragments (independent of this WMMA).
        v16h b1[4];
        if (has_next) {
            #pragma unroll
            for (int kc = 0; kc < 4; ++kc)
                b1[kc] = *(const v16h*)(R16 + boff_n + kc * 32);
        }

        // Mask dwords (streaming read-once -> non-temporal) + ||r||^2.
        int mk[8];
        #pragma unroll
        for (int v = 0; v < 8; ++v)
            mk[v] = __builtin_nontemporal_load(mask + goff + (unsigned)v * rstep);
        const float sr = sqr[soff];

        // 16x16 tile: K = 128 in 4 WMMAs.
        v8f c = {};
        #pragma unroll
        for (int kc = 0; kc < 4; ++kc)
            c = __builtin_amdgcn_wmma_f32_16x16x32_f16(
                    false, a[kc], false, b0[kc], (short)0, c, false, false);

        // d = relu(se+sr-2c); t = keep ? -d : -inf; Z += keep ? exp(-d) : 0
        #pragma unroll
        for (int v = 0; v < 8; ++v) {
            float d = fmaxf(fmaf(-2.0f, c[v], se[v] + sr), 0.0f);
            const bool keep = (mk[v] == 0);
            zacc[v] += keep ? __expf(-d) : 0.0f;
            tout[goff + (unsigned)v * rstep] = keep ? -d : -__builtin_inff();
        }

        if (has_next) {
            #pragma unroll
            for (int kc = 0; kc < 4; ++kc) b0[kc] = b1[kc];
        }
        boff = boff_n;
        goff += 128u;
        soff += 128u;
    }

    // Reduce Z over the 16 column-lanes of each half-wave (rows differ across
    // halves, so never cross the hi boundary).
    #pragma unroll
    for (int sh = 1; sh < 16; sh <<= 1) {
        #pragma unroll
        for (int v = 0; v < 8; ++v)
            zacc[v] += __shfl_xor(zacc[v], sh, 32);
    }

    // Deterministic cross-wave reduction via LDS (fixed summation order).
    __shared__ float sZ[8][16];
    if (lo == 0) {
        #pragma unroll
        for (int v = 0; v < 8; ++v) sZ[wave][hi * 8 + v] = zacc[v];
    }
    __syncthreads();
    if (threadIdx.x < 16) {
        float s = 0.0f;
        #pragma unroll
        for (int w = 0; w < 8; ++w) s += sZ[w][threadIdx.x];
        Zpart[(unsigned)(n0 + threadIdx.x) * M_CHUNKS + mc] = s;
    }
}

// ---------------------------------------------------------------------------
// Kernel 3: Z[row] = sum of M_CHUNKS partials (deterministic order).
// ---------------------------------------------------------------------------
__global__ __launch_bounds__(256) void reduce_z_kernel(
    const float* __restrict__ Zpart, float* __restrict__ Z, int nrows)
{
    const int r = blockIdx.x * blockDim.x + threadIdx.x;
    if (r >= nrows) return;
    float s = 0.0f;
    #pragma unroll
    for (int c = 0; c < M_CHUNKS; ++c) s += Zpart[(unsigned)r * M_CHUNKS + c];
    Z[r] = s;
}

// ---------------------------------------------------------------------------
// Kernel 4: out = t - log(Z[row]); -inf stays -inf. v4f, non-temporal.
// ---------------------------------------------------------------------------
__global__ __launch_bounds__(256) void finalize_kernel(
    float* __restrict__ tout, const float* __restrict__ Z,
    int mshift, unsigned nvec)
{
    const unsigned i = blockIdx.x * blockDim.x + threadIdx.x;
    if (i >= nvec) return;
    const unsigned row = (i << 2) >> mshift;      // M = 2^mshift, M % 4 == 0
    const float lz = __logf(Z[row]);
    v4f* p = (v4f*)tout;
    v4f v = __builtin_nontemporal_load(p + i);
    v -= lz;
    __builtin_nontemporal_store(v, p + i);
}

// ---------------------------------------------------------------------------
extern "C" void kernel_launch(void* const* d_in, const int* in_sizes, int n_in,
                              void* d_out, int out_size, void* d_ws, size_t ws_size,
                              hipStream_t stream)
{
    (void)n_in; (void)out_size; (void)ws_size;

    const float* emb  = (const float*)d_in[0];    // [N, 128]
    const float* refw = (const float*)d_in[1];    // [M, 128]
    const int*   msk  = (const int*)d_in[2];      // [N, M]
    float*       out  = (float*)d_out;            // [N, M]

    const int N = in_sizes[0] / D_DIM;            // 1024
    const int M = in_sizes[1] / D_DIM;            // 32768 (power of two)

    // Workspace layout (bytes)
    char* ws = (char*)d_ws;
    size_t off = 0;
    _Float16* E16  = (_Float16*)(ws + off); off += (size_t)N * D_DIM * 2;  // 256 KB
    _Float16* R16  = (_Float16*)(ws + off); off += (size_t)M * D_DIM * 2;  // 8 MB
    float*    sqe  = (float*)(ws + off);    off += (size_t)N * 4;
    float*    sqr  = (float*)(ws + off);    off += (size_t)M * 4;
    float*    Zprt = (float*)(ws + off);    off += (size_t)N * M_CHUNKS * 4;
    float*    Zrow = (float*)(ws + off);    off += (size_t)N * 4;

    // 1) f16 copies + row norms
    prep_rows_kernel<<<N, 128, 0, stream>>>(emb,  E16, sqe);
    prep_rows_kernel<<<M, 128, 0, stream>>>(refw, R16, sqr);

    // 2) WMMA distance + t-store + Z partials
    const int n_blocks = N / 16;                  // 64
    const int m_cols   = M / M_CHUNKS;            // 1024 (multiple of 128)
    if (M == 32768) {
        dist_gemm_kernel<32768><<<n_blocks * M_CHUNKS, 256, 0, stream>>>(
            E16, R16, sqe, sqr, msk, out, Zprt, M, n_blocks, m_cols);
    } else {
        dist_gemm_kernel<0><<<n_blocks * M_CHUNKS, 256, 0, stream>>>(
            E16, R16, sqe, sqr, msk, out, Zprt, M, n_blocks, m_cols);
    }

    // 3) reduce Z partials
    reduce_z_kernel<<<(N + 255) / 256, 256, 0, stream>>>(Zprt, Zrow, N);

    // 4) out = t - log(Z)
    int mshift = 0;
    while ((1 << mshift) < M) ++mshift;           // M is a power of two
    const unsigned nvec = (unsigned)((size_t)N * M / 4);
    finalize_kernel<<<(int)((nvec + 255) / 256), 256, 0, stream>>>(
        out, Zrow, mshift, nvec);
}
